// MultiHeadAttention4SparseKT_75625784148172
// MI455X (gfx1250) — matrix operand order repaired and checked by reading, not compiled, not run on hardware
//
#include <hip/hip_runtime.h>
#include <hip/hip_bf16.h>

typedef __attribute__((ext_vector_type(16))) __bf16 v16bf;
typedef __attribute__((ext_vector_type(8)))  float  v8f;
typedef __attribute__((ext_vector_type(4)))  unsigned int v4u;
typedef __attribute__((ext_vector_type(8)))  int v8i;
typedef __attribute__((ext_vector_type(4)))  int v4i;

static constexpr int Bn = 8;     // batch
static constexpr int Sn = 1024;  // sequence
static constexpr int DMn = 256;  // model dim
static constexpr int Hn = 8;     // heads
static constexpr int Dn = 32;    // head dim

__device__ __forceinline__ v8f wmma_bf16(v16bf a, v16bf b, v8f c) {
  // D = A(16x32 bf16) * B(32x16 bf16) + C(16x16 f32)
  return __builtin_amdgcn_wmma_f32_16x16x32_bf16(false, a, false, b, (short)0, c,
                                                 false, false);
}

// ---- Tensor Data Mover: DMA a 2-D bf16 panel (dim0 elems wide, dim1 rows,
// row stride stride0 elems) from global memory into LDS at byte offset lds_off.
// D# packed per CDNA5 ISA ch.8: group0 {count=1, lds_addr, global_addr, type=2},
// group1 {data_size=1(2B), tensor_dim0/1, tile_dim0/1, tensor_dim0_stride}.
__device__ __forceinline__ void tdm_load_2d_bf16(unsigned lds_off, const void* gsrc,
                                                 unsigned dim0, unsigned dim1,
                                                 unsigned stride0) {
  unsigned long long ga = (unsigned long long)(uintptr_t)gsrc;
  v4u g0;
  g0[0] = 1u;                                   // count=1, user mode, no gather
  g0[1] = lds_off;                              // lds_addr (bytes)
  g0[2] = (unsigned)(ga & 0xffffffffull);       // global_addr[31:0]
  g0[3] = (unsigned)((ga >> 32) & 0x1ffffffull) // global_addr[56:32]
          | (2u << 30);                         // type = 2 ("image")
  v8i g1;
  g1[0] = (int)(1u << 16);                                            // data_size=1 -> 2B
  g1[1] = (int)((dim0 & 0xffffu) << 16);                              // tensor_dim0[15:0]
  g1[2] = (int)(((dim0 >> 16) & 0xffffu) | ((dim1 & 0xffffu) << 16)); // dim0 hi | dim1 lo
  g1[3] = (int)(((dim1 >> 16) & 0xffffu) | ((dim0 & 0xffffu) << 16)); // dim1 hi | tile_dim0
  g1[4] = (int)(dim1 & 0xffffu);                                      // tile_dim1, tile_dim2=0
  g1[5] = (int)stride0;                                               // tensor_dim0_stride lo
  g1[6] = 0;
  g1[7] = 0;
  v4i z4 = {0, 0, 0, 0};
#if __has_include(<hip/amd_detail/amd_gfx1250_TDM.h>)
  v8i z8 = {0, 0, 0, 0, 0, 0, 0, 0};
  __builtin_amdgcn_tensor_load_to_lds(g0, g1, z4, z4, z8, 0);   // clang-23 / therock: 6 args
#else
  __builtin_amdgcn_tensor_load_to_lds(g0, g1, z4, z4, 0);       // ROCm 7.2 clang-22: 5 args
#endif
}

// A-tile fragment (16x32, bf16), contiguous k order:
//  lanes 0-15: M=lane,    element s: K=s (s=0..15)
//  lanes16-31: M=lane-16, element s: K=16+s
__device__ __forceinline__ v16bf load_a_f32(const float* rowp, int k0, int lane) {
  int kb1 = (lane & 16) ? 16 : 0;
  int kb2 = kb1 + 8;
  v16bf a;
#pragma unroll
  for (int r = 0; r < 4; ++r) {
    a[2 * r]         = (__bf16)rowp[k0 + kb1 + 2 * r];
    a[2 * r + 1]     = (__bf16)rowp[k0 + kb1 + 2 * r + 1];
    a[8 + 2 * r]     = (__bf16)rowp[k0 + kb2 + 2 * r];
    a[8 + 2 * r + 1] = (__bf16)rowp[k0 + kb2 + 2 * r + 1];
  }
  return a;
}

__device__ __forceinline__ v16bf load_a_bf16(const __bf16* rowp, int k0, int lane) {
  int kb1 = (lane & 16) ? 16 : 0;
  int kb2 = kb1 + 8;
  v16bf a;
#pragma unroll
  for (int r = 0; r < 4; ++r) {
    a[2 * r]         = rowp[k0 + kb1 + 2 * r];
    a[2 * r + 1]     = rowp[k0 + kb1 + 2 * r + 1];
    a[8 + 2 * r]     = rowp[k0 + kb2 + 2 * r];
    a[8 + 2 * r + 1] = rowp[k0 + kb2 + 2 * r + 1];
  }
  return a;
}

// ---------------- Kernel 1: Q/K/V projections -> bf16 [B,H,S,D] ----------------
__global__ void proj_qkv_kernel(const float* __restrict__ q,
                                const float* __restrict__ k,
                                const float* __restrict__ v,
                                const float* __restrict__ Wk, const float* __restrict__ bk,
                                const float* __restrict__ Wv, const float* __restrict__ bv,
                                __bf16* __restrict__ QH, __bf16* __restrict__ KH,
                                __bf16* __restrict__ VH) {
  const int lane = threadIdx.x & 31;
  const int wid  = blockIdx.x * (blockDim.x >> 5) + (threadIdx.x >> 5);
  const int TPM  = (Bn * Sn / 16) * (DMn / 16);  // 8192 tiles per matrix
  const int which = wid / TPM;                   // 0=q 1=k 2=v
  const int t  = wid % TPM;
  const int m0 = (t / (DMn / 16)) * 16;
  const int n0 = (t % (DMn / 16)) * 16;

  const float* X    = (which == 0) ? q : (which == 1) ? k : v;
  const float* W    = (which == 2) ? Wv : Wk;
  const float* bias = (which == 2) ? bv : bk;
  __bf16* OUT       = (which == 0) ? QH : (which == 1) ? KH : VH;

  const int row = m0 + (lane & 15);
  const float* xr = X + (size_t)row * DMn;
  const int n = n0 + (lane & 15);
  const float* wr = W + (size_t)n * DMn;
  const int kbB = (lane & 16) ? 16 : 0;

  v8f acc = {};
  for (int k0 = 0; k0 < DMn; k0 += 32) {
    v16bf a = load_a_f32(xr, k0, lane);
    v16bf bm;
#pragma unroll
    for (int jj = 0; jj < 16; ++jj) bm[jj] = (__bf16)wr[k0 + kbB + jj];
    acc = wmma_bf16(a, bm, acc);
  }

  const float bv_ = bias[n];
  const int h = n >> 5, d = n & 31;
#pragma unroll
  for (int r = 0; r < 8; ++r) {
    const int m  = m0 + r + ((lane & 16) ? 8 : 0);
    const int bb = m >> 10, ss = m & 1023;
    OUT[(((size_t)bb * Hn + h) * Sn + ss) * Dn + d] = (__bf16)(acc[r] + bv_);
  }
}

// ---------- Kernel 2: scores + softmax + top-k sparsify + zero_pad -> attn ----------
// One block (256 thr) per (b,h, 16-row strip).
// Dynamic LDS: [0,64K) TDM-staged KH panel; [64K,128K) score strip; + 2K scratch.
__global__ void attn_kernel(const __bf16* __restrict__ QH, const __bf16* __restrict__ KH,
                            const int* __restrict__ zero_pad_p,
                            const int* __restrict__ k_index_p,
                            float* __restrict__ attnOut) {
  extern __shared__ char smem_raw[];
  __bf16* khl = (__bf16*)smem_raw;              // [1024][32] staged K panel
  float* sc   = (float*)(smem_raw + 65536);     // [16][1024]
  float* red  = sc + 16 * 1024;                 // [16][16]
  float* redc = red + 256;                      // [16][16]

  const int lane = threadIdx.x & 31;
  const int wave = threadIdx.x >> 5;
  const int bh = blockIdx.x / (Sn / 16);
  const int i0 = (blockIdx.x % (Sn / 16)) * 16;

  // ---- stage K panel for this (b,h) into LDS via Tensor Data Mover
  if (wave == 0) {
    tdm_load_2d_bf16(0u, KH + (size_t)bh * Sn * Dn, (unsigned)Dn, (unsigned)Sn,
                     (unsigned)Dn);
    __builtin_amdgcn_s_wait_tensorcnt(0);
  }
  __syncthreads();

  // ---- phase 1: scores via WMMA (K = D = 32, single wmma per 16x16 tile)
  {
    const int i = i0 + (lane & 15);
    const __bf16* qrow = QH + ((size_t)bh * Sn + i) * Dn;
    v16bf a = load_a_bf16(qrow, 0, lane);
    const int kbB = (lane & 16) ? 16 : 0;
    const float scale = 0.17677669529663687f;  // 1/sqrt(32)
    for (int ct = wave; ct < Sn / 16; ct += 8) {
      const int j0 = ct * 16;
      const int j = j0 + (lane & 15);
      const __bf16* krow = khl + (size_t)j * Dn + kbB;  // LDS reads
      v16bf bm;
#pragma unroll
      for (int jj = 0; jj < 16; ++jj) bm[jj] = krow[jj];
      v8f accv = {};
      accv = wmma_bf16(a, bm, accv);
#pragma unroll
      for (int r = 0; r < 8; ++r) {
        const int ii = i0 + r + ((lane & 16) ? 8 : 0);
        float s = accv[r] * scale;
        if (j > ii) s = -1e32f;  // causal mask
        sc[(ii - i0) * 1024 + j] = s;
      }
    }
  }
  __syncthreads();

  // ---- phase 2: row softmax (16 threads per row, each owns j = sub + 16*m)
  const int r = threadIdx.x >> 4;
  const int sub = threadIdx.x & 15;
  float* srow = sc + r * 1024;

  float lmax = -3.0e38f;
  for (int m = 0; m < 64; ++m) lmax = fmaxf(lmax, srow[sub + (m << 4)]);
  red[r * 16 + sub] = lmax;
  __syncthreads();
  float rowmax = -3.0e38f;
  for (int tt = 0; tt < 16; ++tt) rowmax = fmaxf(rowmax, red[r * 16 + tt]);
  __syncthreads();
  float lsum = 0.f;
  for (int m = 0; m < 64; ++m) lsum += __expf(srow[sub + (m << 4)] - rowmax);
  red[r * 16 + sub] = lsum;
  __syncthreads();
  float rowsum = 0.f;
  for (int tt = 0; tt < 16; ++tt) rowsum += red[r * 16 + tt];
  const float inv = 1.0f / rowsum;
  __syncthreads();
  for (int m = 0; m < 64; ++m) {
    const int j = sub + (m << 4);
    srow[j] = __expf(srow[j] - rowmax) * inv;
  }
  __syncthreads();

  // ---- phase 3: tie-correct top-k threshold (k-th largest incl. duplicates)
  const int kidx = *k_index_p;
  const int zp = *zero_pad_p;
  const int gi = i0 + r;
  const bool sparsify = (kidx + 1 < Sn) && (gi > kidx);

  float cur = 3.4e38f, cnt = 0.f, topmax = -3.4e38f;
  for (int pass = 0; pass < kidx; ++pass) {  // uniform trip count -> legal barriers
    float lmax2 = -3.4e38f;
    if (cnt < (float)kidx) {
      for (int m = 0; m < 64; ++m) {
        const float x = srow[sub + (m << 4)];
        if (x < cur) lmax2 = fmaxf(lmax2, x);
      }
    }
    red[r * 16 + sub] = lmax2;
    __syncthreads();
    float mx = -3.4e38f;
    for (int tt = 0; tt < 16; ++tt) mx = fmaxf(mx, red[r * 16 + tt]);
    float lc = 0.f;
    if (cnt < (float)kidx) {
      for (int m = 0; m < 64; ++m)
        if (srow[sub + (m << 4)] == mx) lc += 1.f;
    }
    redc[r * 16 + sub] = lc;
    __syncthreads();
    float c = 0.f;
    for (int tt = 0; tt < 16; ++tt) c += redc[r * 16 + tt];
    if (cnt < (float)kidx) {
      cnt += c;
      cur = mx;
      if (pass == 0) topmax = mx;
    }
    __syncthreads();
  }
  const float thresh = cur;

  // re-softmax over kept entries (non-kept -> exp(-1e16-max) == 0 in fp32)
  float lsum2 = 0.f;
  if (sparsify) {
    for (int m = 0; m < 64; ++m) {
      const float x = srow[sub + (m << 4)];
      if (x >= thresh) lsum2 += __expf(x - topmax);
    }
  }
  red[r * 16 + sub] = lsum2;
  __syncthreads();
  float rsk = 0.f;
  for (int tt = 0; tt < 16; ++tt) rsk += red[r * 16 + tt];
  const float invk = 1.0f / rsk;
  if (sparsify) {
    for (int m = 0; m < 64; ++m) {
      const int j = sub + (m << 4);
      const float x = srow[j];
      srow[j] = (x >= thresh) ? __expf(x - topmax) * invk : 0.0f;
    }
  }
  __syncthreads();

  // ---- phase 4: write attn (with zero_pad on global row 0)
  const size_t base = ((size_t)bh * Sn + gi) * Sn;
  const bool zrow = (zp != 0) && (gi == 0);
  for (int m = 0; m < 64; ++m) {
    const int j = sub + (m << 4);
    attnOut[base + j] = zrow ? 0.0f : srow[j];
  }
}

// ------- Kernel 3: out_heads = attn @ V -> bf16 concat [B,S,DM] -------
// Blocks of 8 waves share one (b,h); V panel TDM-staged to LDS (strided
// B-fragment gathers become conflict-free LDS reads).
__global__ void av_kernel(const float* __restrict__ attnOut, const __bf16* __restrict__ VH,
                          __bf16* __restrict__ concat) {
  extern __shared__ char smem_raw[];
  __bf16* vhl = (__bf16*)smem_raw;  // [1024][32] staged V panel (64KB)

  const int lane = threadIdx.x & 31;
  const int wave = threadIdx.x >> 5;
  const int wid = blockIdx.x * (blockDim.x >> 5) + (threadIdx.x >> 5);
  const int bh = wid >> 7;            // uniform within a block (128 tiles per head)
  const int t = wid & 127;
  const int m0 = (t >> 1) * 16;
  const int n0 = (t & 1) * 16;
  const int b = bh >> 3, h = bh & 7;

  if (wave == 0) {
    tdm_load_2d_bf16(0u, VH + (size_t)bh * Sn * Dn, (unsigned)Dn, (unsigned)Sn,
                     (unsigned)Dn);
    __builtin_amdgcn_s_wait_tensorcnt(0);
  }
  __syncthreads();

  const int i = m0 + (lane & 15);
  const float* arow = attnOut + ((size_t)bh * Sn + i) * Sn;
  const int d = n0 + (lane & 15);
  const int kbB = (lane & 16) ? 16 : 0;

  v8f acc = {};
  for (int j0 = 0; j0 < Sn; j0 += 32) {
    __builtin_prefetch(arow + j0 + 128, 0, 1);  // stream next attn chunk
    v16bf a = load_a_f32(arow, j0, lane);
    v16bf bm;
#pragma unroll
    for (int jj = 0; jj < 16; ++jj)
      bm[jj] = vhl[(size_t)(j0 + kbB + jj) * Dn + d];
    acc = wmma_bf16(a, bm, acc);
  }
#pragma unroll
  for (int rr = 0; rr < 8; ++rr) {
    const int ii = m0 + rr + ((lane & 16) ? 8 : 0);
    concat[((size_t)b * Sn + ii) * DMn + h * Dn + d] = (__bf16)acc[rr];
  }
}

// ---------------- Kernel 4: output = concat @ Wo^T + bo (fp32) ----------------
__global__ void out_proj_kernel(const __bf16* __restrict__ concat,
                                const float* __restrict__ Wo, const float* __restrict__ bo,
                                float* __restrict__ out) {
  const int lane = threadIdx.x & 31;
  const int wid = blockIdx.x * (blockDim.x >> 5) + (threadIdx.x >> 5);
  const int m0 = (wid / (DMn / 16)) * 16;
  const int n0 = (wid % (DMn / 16)) * 16;

  const int row = m0 + (lane & 15);
  const __bf16* ar = concat + (size_t)row * DMn;
  const int n = n0 + (lane & 15);
  const float* wr = Wo + (size_t)n * DMn;
  const int kbB = (lane & 16) ? 16 : 0;

  v8f acc = {};
  for (int k0 = 0; k0 < DMn; k0 += 32) {
    v16bf a = load_a_bf16(ar, k0, lane);
    v16bf bm;
#pragma unroll
    for (int jj = 0; jj < 16; ++jj) bm[jj] = (__bf16)wr[k0 + kbB + jj];
    acc = wmma_bf16(a, bm, acc);
  }
  const float bias = bo[n];
#pragma unroll
  for (int rr = 0; rr < 8; ++rr) {
    const int m = m0 + rr + ((lane & 16) ? 8 : 0);
    out[(size_t)m * DMn + n0 + (lane & 15)] = acc[rr] + bias;
  }
}

extern "C" void kernel_launch(void* const* d_in, const int* in_sizes, int n_in,
                              void* d_out, int out_size, void* d_ws, size_t ws_size,
                              hipStream_t stream) {
  const float* q  = (const float*)d_in[0];
  const float* k  = (const float*)d_in[1];
  const float* v  = (const float*)d_in[2];
  // d_in[3] = mask (causal tril) — implied by j<=i in-kernel
  const int* zp   = (const int*)d_in[4];
  const int* ki   = (const int*)d_in[5];
  const float* Wk = (const float*)d_in[6];
  const float* bk = (const float*)d_in[7];
  const float* Wv = (const float*)d_in[8];
  const float* bv = (const float*)d_in[9];
  const float* Wo = (const float*)d_in[10];
  const float* bo = (const float*)d_in[11];

  float* out  = (float*)d_out;                       // [B,S,DM]
  float* attn = out + (size_t)Bn * Sn * DMn;         // [B,H,S,S]

  const size_t headElems = (size_t)Bn * Hn * Sn * Dn;  // 2,097,152
  __bf16* QH = (__bf16*)d_ws;
  __bf16* KH = QH + headElems;
  __bf16* VH = KH + headElems;
  __bf16* concat = VH + headElems;                   // [B,S,DM] bf16

  dim3 blk(256);

  // 3 matrices * 512 row-tiles * 16 col-tiles = 24576 wave-tiles / 8 waves
  proj_qkv_kernel<<<dim3(24576 / 8), blk, 0, stream>>>(q, k, v, Wk, bk, Wv, bv, QH, KH, VH);

  // one block per (b,h,16-row strip): LDS = 64K (K panel) + 64K (scores) + 2K
  const size_t shmem_attn = 65536 + (size_t)(16 * 1024 + 2 * 256) * sizeof(float);
  attn_kernel<<<dim3(Bn * Hn * (Sn / 16)), blk, shmem_attn, stream>>>(QH, KH, zp, ki, attn);

  // 64 (b,h) * 64 row-tiles * 2 col-tiles = 8192 wave-tiles / 8 waves; 64K V panel
  av_kernel<<<dim3(8192 / 8), blk, 65536, stream>>>(attn, VH, concat);

  // 512 row-tiles * 16 col-tiles = 8192 wave-tiles / 8 waves
  out_proj_kernel<<<dim3(8192 / 8), blk, 0, stream>>>(concat, Wo, bo, out);

  (void)in_sizes; (void)n_in; (void)out_size; (void)ws_size;
}
